// CausalSelfAttention_27917287424522
// MI455X (gfx1250) — matrix operand order repaired and checked by reading, not compiled, not run on hardware
//
#include <hip/hip_runtime.h>
#include <hip/hip_bf16.h>

#define BB   4
#define LL   2048
#define DD   768
#define HH   6
#define HDIM 128

typedef __bf16 bf16;
typedef __bf16 v16bf __attribute__((ext_vector_type(16)));
typedef float  v8f   __attribute__((ext_vector_type(8)));

#if defined(__gfx1250__) && __has_builtin(__builtin_amdgcn_tensor_load_to_lds) && \
    __has_builtin(__builtin_amdgcn_s_wait_tensorcnt)
#define HAVE_TDM 1
#else
#define HAVE_TDM 0
#endif

static __device__ __forceinline__ v8f zero8() {
  v8f z;
#pragma unroll
  for (int i = 0; i < 8; ++i) z[i] = 0.0f;
  return z;
}

static __device__ __forceinline__ v8f wmma_bf16(v16bf a, v16bf b, v8f c) {
  // 8 args: (neg_a, A, neg_b, B, c_mod, C, reuse_a, reuse_b)
  return __builtin_amdgcn_wmma_f32_16x16x32_bf16(false, a, false, b, (short)0, c,
                                                 false, false);
}

#if HAVE_TDM
typedef unsigned tdm_u32x4 __attribute__((ext_vector_type(4)));
typedef int      tdm_i32x8 __attribute__((ext_vector_type(8)));
typedef int      tdm_i32x4 __attribute__((ext_vector_type(4)));

// Issue one TDM load: 2-D tile (td0 x td1 elements, bf16) from global (row
// stride = stride0 elements) into LDS at byte offset lds_off, rows packed.
// td1==0 encodes a 1-D tile (tile_dim1 "unused").  D# layout per ISA ch.8.
static __device__ __forceinline__ void tdm_load_2d(unsigned lds_off, const void* gptr,
                                                   unsigned td0, unsigned td1,
                                                   unsigned stride0) {
  const unsigned long long ga = (unsigned long long)gptr;
  const unsigned tensor_d0 = td0;
  const unsigned tensor_d1 = td1 ? td1 : 1u;
  tdm_u32x4 g0;
  g0[0] = 1u;                                           // count=1, user mode
  g0[1] = lds_off;                                      // lds_addr (bytes)
  g0[2] = (unsigned)(ga & 0xffffffffu);                 // global_addr[31:0]
  g0[3] = (unsigned)((ga >> 32) & 0x1ffffffu) | (2u << 30);  // addr[56:32]|type=2
  tdm_i32x8 g1;
  g1[0] = (int)(1u << 16);                              // data_size=2B, wg_mask=0
  g1[1] = (int)((tensor_d0 & 0xffffu) << 16);           // tensor_dim0[15:0]
  g1[2] = (int)(((tensor_d0 >> 16) & 0xffffu) | ((tensor_d1 & 0xffffu) << 16));
  g1[3] = (int)(((tensor_d1 >> 16) & 0xffffu) | ((td0 & 0xffffu) << 16)); // tile_dim0
  g1[4] = (int)(td1 & 0xffffu);                         // tile_dim1 (0=1-D), tile_dim2=0
  g1[5] = (int)stride0;                                 // tensor_dim0_stride[31:0]
  g1[6] = 0;                                            // stride0 hi / dim1_stride lo
  g1[7] = 0;
  tdm_i32x4 g2;                                         // dims 2/3 benign (=1)
  g2[0] = 1; g2[1] = 1; g2[2] = 0; g2[3] = 0;
  tdm_i32x4 g3;                                         // dim4=1, tile_dim4 unused
  g3[0] = 0; g3[1] = (int)(1u << 16); g3[2] = 0; g3[3] = 0;
  tdm_i32x8 g4;                                         // uncharacterized extra group
#pragma unroll
  for (int i = 0; i < 8; ++i) g4[i] = 0;
  __builtin_amdgcn_tensor_load_to_lds(g0, g1, g2, g3, g4, 0);
}
#endif

// ---- Fragment loaders (CDNA5 ISA 7.12.2, wave32) -------------------------
static __device__ __forceinline__ v16bf load_a_frag(const bf16* base, int lda, int lane) {
  const int hf = (lane >> 4) & 1;
  const int m  = lane & 15;
  v16bf a;
#pragma unroll
  for (int v = 0; v < 8; ++v) {
    int k = (hf << 4) + (v << 1);
    a[2 * v]     = base[m * lda + k];
    a[2 * v + 1] = base[m * lda + k + 1];
  }
  return a;
}

static __device__ __forceinline__ v16bf load_b_frag_nk(const bf16* base, int ldb, int lane) {
  const int hf = (lane >> 4) & 1;
  const int n  = lane & 15;
  v16bf b;
#pragma unroll
  for (int v = 0; v < 8; ++v) {
    int k = (hf << 4) + (v << 1);
    b[2 * v]     = base[n * ldb + k];
    b[2 * v + 1] = base[n * ldb + k + 1];
  }
  return b;
}

static __device__ __forceinline__ v16bf load_b_frag_kn(const bf16* base, int ldb, int lane) {
  const int hf = (lane >> 4) & 1;
  const int n  = lane & 15;
  v16bf b;
#pragma unroll
  for (int v = 0; v < 8; ++v) {
    int k = (hf << 4) + (v << 1);
    b[2 * v]     = base[k * ldb + n];
    b[2 * v + 1] = base[(k + 1) * ldb + n];
  }
  return b;
}

// ---- Conversion kernels ---------------------------------------------------
__global__ void k_cvt_x(const float* __restrict__ x, bf16* __restrict__ xb) {
  size_t i = ((size_t)blockIdx.x * blockDim.x + threadIdx.x) * 4;
  float4 v = *(const float4*)(x + i);
  bf16* p = xb + i;
  p[0] = (bf16)v.x; p[1] = (bf16)v.y; p[2] = (bf16)v.z; p[3] = (bf16)v.w;
}

__global__ void k_cvt_w(const float* W0, const float* W1, const float* W2, const float* W3,
                        bf16* T0, bf16* T1, bf16* T2, bf16* T3) {
  __shared__ float tile[32][33];
  const float* W = (blockIdx.z == 0) ? W0 : (blockIdx.z == 1) ? W1 : (blockIdx.z == 2) ? W2 : W3;
  bf16*        T = (blockIdx.z == 0) ? T0 : (blockIdx.z == 1) ? T1 : (blockIdx.z == 2) ? T2 : T3;
  int k0 = blockIdx.x * 32, n0 = blockIdx.y * 32;
  int tx = threadIdx.x, ty = threadIdx.y;
#pragma unroll
  for (int yy = 0; yy < 32; yy += 8)
    tile[ty + yy][tx] = W[(size_t)(k0 + ty + yy) * DD + n0 + tx];
  __syncthreads();
#pragma unroll
  for (int yy = 0; yy < 32; yy += 8)
    T[(size_t)(n0 + ty + yy) * DD + k0 + tx] = (bf16)tile[tx][ty + yy];
}

// ---- Shared 128x128 tile GEMM core (C = A[M,K] * Bt[N,K]^T) --------------
// 8 waves in 4(wm) x 2(wn); each wave owns 2x4 16x16 tiles.  Double-buffered
// LDS; TDM DMA overlaps the whole compute of the current K-slice.
static __device__ __forceinline__ void gemm_core_128x128(
    const bf16* __restrict__ A, int lda, int mbase,
    const bf16* __restrict__ Bt, int ldb, int nbase,
    int kdim, bf16* lA, bf16* lB, v8f acc[2][4]) {
  const int tid = threadIdx.x;
  const int wave = tid >> 5, lane = tid & 31;
  const int wm = wave >> 1, wn = wave & 1;
#pragma unroll
  for (int i = 0; i < 2; ++i)
#pragma unroll
    for (int j = 0; j < 4; ++j) acc[i][j] = zero8();

  const int nk = kdim >> 5;
#if HAVE_TDM
  const bool issuer = (wave == 0);
  const unsigned offA = (unsigned)(unsigned long long)(const void*)lA;
  const unsigned offB = (unsigned)(unsigned long long)(const void*)lB;
  if (issuer) {
    tdm_load_2d(offA, A + (size_t)mbase * lda, 32, 128, (unsigned)lda);
    tdm_load_2d(offB, Bt + (size_t)nbase * ldb, 32, 128, (unsigned)ldb);
  }
#endif
  for (int ki = 0; ki < nk; ++ki) {
    const int cur = ki & 1;
#if HAVE_TDM
    if (issuer) __builtin_amdgcn_s_wait_tensorcnt(0);
    __syncthreads();                    // publish tile ki; all waves done with buf
    if (issuer && (ki + 1) < nk) {
      const unsigned nxt = (unsigned)(cur ^ 1);
      tdm_load_2d(offA + nxt * 8192u, A + (size_t)mbase * lda + (ki + 1) * 32, 32, 128,
                  (unsigned)lda);
      tdm_load_2d(offB + nxt * 8192u, Bt + (size_t)nbase * ldb + (ki + 1) * 32, 32, 128,
                  (unsigned)ldb);
    }
#else
    __syncthreads();
    {
      const int k0 = ki * 32;
      unsigned* sA = (unsigned*)(lA + cur * 4096);
      unsigned* sB = (unsigned*)(lB + cur * 4096);
#pragma unroll
      for (int i = 0; i < 8; ++i) {
        int p = tid + i * 256;
        int row = p >> 4, col = p & 15;
        sA[p] = *(const unsigned*)(A + (size_t)(mbase + row) * lda + k0 + col * 2);
        sB[p] = *(const unsigned*)(Bt + (size_t)(nbase + row) * ldb + k0 + col * 2);
      }
    }
    __syncthreads();
#endif
    const bf16* cA = lA + cur * 4096;
    const bf16* cB = lB + cur * 4096;
    v16bf af[2];
#pragma unroll
    for (int ms = 0; ms < 2; ++ms)
      af[ms] = load_a_frag(cA + (wm * 32 + ms * 16) * 32, 32, lane);
#pragma unroll
    for (int ns = 0; ns < 4; ++ns) {
      v16bf bfm = load_b_frag_nk(cB + (wn * 64 + ns * 16) * 32, 32, lane);
#pragma unroll
      for (int ms = 0; ms < 2; ++ms) acc[ms][ns] = wmma_bf16(af[ms], bfm, acc[ms][ns]);
    }
  }
}

// ---- QKV projection: out = (x @ W + b) [*scale for Q], bf16 [b,h,l,d] ----
__global__ __launch_bounds__(256) void k_gemm_qkv(
    const bf16* __restrict__ xb,
    const bf16* __restrict__ WtQ, const bf16* __restrict__ WtK, const bf16* __restrict__ WtV,
    const float* __restrict__ bq, const float* __restrict__ bk, const float* __restrict__ bv,
    bf16* __restrict__ Q, bf16* __restrict__ K, bf16* __restrict__ V) {
  __shared__ bf16 lA[2 * 128 * 32];
  __shared__ bf16 lB[2 * 128 * 32];
  const int mode = blockIdx.z;
  const bf16* Bt    = (mode == 0) ? WtQ : (mode == 1) ? WtK : WtV;
  const float* bias = (mode == 0) ? bq : (mode == 1) ? bk : bv;
  bf16* Out = (mode == 0) ? Q : (mode == 1) ? K : V;
  const float scale = (mode == 0) ? 0.08838834764831845f : 1.0f;  // 1/sqrt(128)
  const int mbase = blockIdx.x * 128, nbase = blockIdx.y * 128;
  v8f acc[2][4];
  gemm_core_128x128(xb, DD, mbase, Bt, DD, nbase, DD, lA, lB, acc);
  const int tid = threadIdx.x, wave = tid >> 5, lane = tid & 31;
  const int wm = wave >> 1, wn = wave & 1, hf = (lane >> 4) & 1, n = lane & 15;
#pragma unroll
  for (int ms = 0; ms < 2; ++ms)
#pragma unroll
    for (int ns = 0; ns < 4; ++ns)
#pragma unroll
      for (int r = 0; r < 8; ++r) {
        int gm = mbase + wm * 32 + ms * 16 + r + 8 * hf;   // C/D layout: M = r + 8*half
        int gn = nbase + wn * 64 + ns * 16 + n;
        float c = (acc[ms][ns][r] + bias[gn]) * scale;
        int b = gm >> 11, l = gm & 2047;
        int h = gn >> 7,  d = gn & 127;
        Out[(((size_t)b * HH + h) * LL + l) * HDIM + d] = (bf16)c;
      }
}

// ---- Flash attention: per (qtile,h,b); 8 waves x 16 query rows -----------
__global__ __launch_bounds__(256) void k_attn(
    const bf16* __restrict__ Q, const bf16* __restrict__ K,
    const bf16* __restrict__ V, bf16* __restrict__ Ob) {
  __shared__ bf16 Kt[2][128 * HDIM];    // 2 x 32 KB key chunks
  __shared__ bf16 Vt[2][128 * HDIM];    // 2 x 32 KB value chunks
  __shared__ bf16 Pt[8][16 * 128];      // 32 KB wave-private P tiles
  const int qt = blockIdx.x, h = blockIdx.y, b = blockIdx.z;
  const int tid = threadIdx.x, wave = tid >> 5, lane = tid & 31;
  const int hf = (lane >> 4) & 1, ln = lane & 15;
  const size_t bh = (((size_t)b * HH) + h) * (size_t)LL * HDIM;
  const int qb = qt * 128;
  const bf16* Qp = Q + bh + (size_t)qb * HDIM;

  v16bf qf[4];                          // this wave's 16x128 Q tile (pre-scaled)
#pragma unroll
  for (int kt = 0; kt < 4; ++kt)
    qf[kt] = load_a_frag(Qp + (wave * 16) * HDIM + kt * 32, HDIM, lane);

  v8f Oacc[8];
#pragma unroll
  for (int nt = 0; nt < 8; ++nt) Oacc[nt] = zero8();
  float mrow[8], lrow[8];
#pragma unroll
  for (int r = 0; r < 8; ++r) { mrow[r] = -1e30f; lrow[r] = 0.0f; }

  int nchunks = qt + 2;                 // tril(k=1): keys up to qb+127+1
  if (nchunks > LL / 128) nchunks = LL / 128;

#if HAVE_TDM
  const bool issuer = (wave == 0);
  const unsigned offK = (unsigned)(unsigned long long)(const void*)&Kt[0][0];
  const unsigned offV = (unsigned)(unsigned long long)(const void*)&Vt[0][0];
  if (issuer) {                         // prologue: chunk 0 into buffer 0
    tdm_load_2d(offK, K + bh, 16384, 0, 0);
    tdm_load_2d(offV, V + bh, 16384, 0, 0);
  }
#endif

  for (int c = 0; c < nchunks; ++c) {
    const int cur = c & 1;
#if HAVE_TDM
    if (issuer) __builtin_amdgcn_s_wait_tensorcnt(0);
    __syncthreads();                    // publish chunk c; prior buf reads done
    if (issuer && (c + 1) < nchunks) {
      const unsigned nxt = (unsigned)(cur ^ 1);
      tdm_load_2d(offK + nxt * 32768u, K + bh + (size_t)(c + 1) * 128 * HDIM, 16384, 0, 0);
      tdm_load_2d(offV + nxt * 32768u, V + bh + (size_t)(c + 1) * 128 * HDIM, 16384, 0, 0);
    }
#else
    __syncthreads();
    {
      const uint4* gk = (const uint4*)(K + bh + (size_t)c * 128 * HDIM);
      const uint4* gv = (const uint4*)(V + bh + (size_t)c * 128 * HDIM);
      uint4* sk = (uint4*)&Kt[cur][0];
      uint4* sv = (uint4*)&Vt[cur][0];
#pragma unroll
      for (int i = 0; i < 8; ++i) {
        sk[tid + i * 256] = gk[tid + i * 256];
        sv[tid + i * 256] = gv[tid + i * 256];
      }
    }
    __syncthreads();
#endif

    v8f S[8];                           // 16 x 128 scores (f32)
#pragma unroll
    for (int t = 0; t < 8; ++t) S[t] = zero8();
#pragma unroll
    for (int kt = 0; kt < 4; ++kt)
#pragma unroll
      for (int t = 0; t < 8; ++t) {
        v16bf bk_ = load_b_frag_nk(&Kt[cur][(t * 16) * HDIM + kt * 32], HDIM, lane);
        S[t] = wmma_bf16(qf[kt], bk_, S[t]);
      }

    const int kc = c * 128;
    float alpha[8];
#pragma unroll
    for (int r = 0; r < 8; ++r) {
      int qi = qb + wave * 16 + r + 8 * hf;
      float mx = -1e30f;
#pragma unroll
      for (int t = 0; t < 8; ++t) {
        int kj = kc + t * 16 + ln;
        float s = (kj <= qi + 1) ? S[t][r] : -__builtin_inff();
        S[t][r] = s;
        mx = fmaxf(mx, s);
      }
#pragma unroll
      for (int off = 1; off < 16; off <<= 1) mx = fmaxf(mx, __shfl_xor(mx, off, 32));
      float mnew = fmaxf(mrow[r], mx);
      float a = __expf(mrow[r] - mnew);
      alpha[r] = a;
      float rs = 0.0f;
#pragma unroll
      for (int t = 0; t < 8; ++t) { float p = __expf(S[t][r] - mnew); S[t][r] = p; rs += p; }
#pragma unroll
      for (int off = 1; off < 16; off <<= 1) rs += __shfl_xor(rs, off, 32);
      lrow[r] = lrow[r] * a + rs;
      mrow[r] = mnew;
    }
#pragma unroll
    for (int nt = 0; nt < 8; ++nt)
#pragma unroll
      for (int r = 0; r < 8; ++r) Oacc[nt][r] *= alpha[r];

    bf16* myP = &Pt[wave][0];           // re-shape P: C/D layout -> A layout
#pragma unroll
    for (int t = 0; t < 8; ++t)
#pragma unroll
      for (int r = 0; r < 8; ++r)
        myP[(r + 8 * hf) * 128 + t * 16 + ln] = (bf16)S[t][r];
    __syncthreads();

#pragma unroll
    for (int kt2 = 0; kt2 < 4; ++kt2) {
      v16bf ap = load_a_frag(myP + kt2 * 32, 128, lane);
#pragma unroll
      for (int nt = 0; nt < 8; ++nt) {
        v16bf bv_ = load_b_frag_kn(&Vt[cur][(kt2 * 32) * HDIM + nt * 16], HDIM, lane);
        Oacc[nt] = wmma_bf16(ap, bv_, Oacc[nt]);
      }
    }
  }

  float rl[8];
#pragma unroll
  for (int r = 0; r < 8; ++r) rl[r] = 1.0f / lrow[r];
  // O layout: [b, l, h*128+d] so projection is a plain [8192,768] GEMM
  const size_t orow0 = ((size_t)b * LL + qb + wave * 16) * DD + (size_t)h * HDIM;
#pragma unroll
  for (int nt = 0; nt < 8; ++nt)
#pragma unroll
    for (int r = 0; r < 8; ++r) {
      int m = r + 8 * hf;
      Ob[orow0 + (size_t)m * DD + nt * 16 + ln] = (bf16)(Oacc[nt][r] * rl[r]);
    }
}

// ---- Output projection: out = O @ Wo + bo (f32) ---------------------------
__global__ __launch_bounds__(256) void k_gemm_proj(
    const bf16* __restrict__ Ob, const bf16* __restrict__ WtO,
    const float* __restrict__ bo, float* __restrict__ out) {
  __shared__ bf16 lA[2 * 128 * 32];
  __shared__ bf16 lB[2 * 128 * 32];
  const int mbase = blockIdx.x * 128, nbase = blockIdx.y * 128;
  v8f acc[2][4];
  gemm_core_128x128(Ob, DD, mbase, WtO, DD, nbase, DD, lA, lB, acc);
  const int tid = threadIdx.x, wave = tid >> 5, lane = tid & 31;
  const int wm = wave >> 1, wn = wave & 1, hf = (lane >> 4) & 1, n = lane & 15;
#pragma unroll
  for (int ms = 0; ms < 2; ++ms)
#pragma unroll
    for (int ns = 0; ns < 4; ++ns)
#pragma unroll
      for (int r = 0; r < 8; ++r) {
        int gm = mbase + wm * 32 + ms * 16 + r + 8 * hf;
        int gn = nbase + wn * 64 + ns * 16 + n;
        out[(size_t)gm * DD + gn] = acc[ms][ns][r] + bo[gn];
      }
}

extern "C" void kernel_launch(void* const* d_in, const int* in_sizes, int n_in,
                              void* d_out, int out_size, void* d_ws, size_t ws_size,
                              hipStream_t stream) {
  (void)in_sizes; (void)n_in; (void)out_size; (void)ws_size;
  const float* x  = (const float*)d_in[0];
  const float* Wq = (const float*)d_in[1];
  const float* bq = (const float*)d_in[2];
  const float* Wk = (const float*)d_in[3];
  const float* bk = (const float*)d_in[4];
  const float* Wv = (const float*)d_in[5];
  const float* bv = (const float*)d_in[6];
  const float* Wo = (const float*)d_in[7];
  const float* bo = (const float*)d_in[8];
  float* out = (float*)d_out;

  const size_t NX = (size_t)BB * LL * DD;   // 6291456 elements
  const size_t NW = (size_t)DD * DD;        // 589824 elements
  bf16* xb  = (bf16*)d_ws;                  // total ws use ~64.5 MB (bf16)
  bf16* WtQ = xb  + NX;
  bf16* WtK = WtQ + NW;
  bf16* WtV = WtK + NW;
  bf16* WtO = WtV + NW;
  bf16* Qb  = WtO + NW;
  bf16* Kb  = Qb + NX;
  bf16* Vb  = Kb + NX;
  bf16* Ob  = Vb + NX;

  k_cvt_x<<<(unsigned)(NX / (256 * 4)), 256, 0, stream>>>(x, xb);
  k_cvt_w<<<dim3(DD / 32, DD / 32, 4), dim3(32, 8), 0, stream>>>(
      Wq, Wk, Wv, Wo, WtQ, WtK, WtV, WtO);
  k_gemm_qkv<<<dim3(BB * LL / 128, DD / 128, 3), 256, 0, stream>>>(
      xb, WtQ, WtK, WtV, bq, bk, bv, Qb, Kb, Vb);
  k_attn<<<dim3(LL / 128, HH, BB), 256, 0, stream>>>(Qb, Kb, Vb, Ob);
  k_gemm_proj<<<dim3(BB * LL / 128, DD / 128, 1), 256, 0, stream>>>(Ob, WtO, bo, out);
}
